// AdaptiveFeaturePropagation_47708496724547
// MI455X (gfx1250) — matrix operand order repaired and checked by reading, not compiled, not run on hardware
//
#include <hip/hip_runtime.h>
#include <hip/hip_bf16.h>

// ---------------------------------------------------------------------------
// CDNA5 / gfx1250 implementation.
//  - 3x3 convs: tap-major implicit GEMM, v_wmma_f32_16x16x32_bf16 (f32 accum)
//    * weights pre-repacked to [tap][Cout][Cin] bf16 -> contiguous A tiles
//    * activations pre-converted to bf16; branchless boundary masking
//    * K=64 per stage (8 WMMAs), double-buffered LDS, 1 barrier/stage,
//      flattened (tap,cin) chunk stream for cross-tap pipelining
//  - 1x1 conv + softmax: fused VALU kernel (scalarized uniform weight loads)
//  - spatially-variant conv: LDS-cached per-pixel kernels, streaming FMA
// ---------------------------------------------------------------------------

typedef __attribute__((ext_vector_type(16))) __bf16 v16bf;
typedef __attribute__((ext_vector_type(8)))  float  v8f;

__device__ __forceinline__ unsigned short f32_to_bf16(float f) {
    unsigned u = __float_as_uint(f);
    u += 0x7fffu + ((u >> 16) & 1u);   // round-to-nearest-even
    return (unsigned short)(u >> 16);
}

// ---------------- fp32 -> bf16 streaming convert (8 elems/thread) ----------
__global__ __launch_bounds__(256)
void cvt_f32_bf16(const float* __restrict__ in, unsigned short* __restrict__ out)
{
    size_t i = ((size_t)blockIdx.x * 256 + threadIdx.x) * 8;
    float4 a = *(const float4*)(in + i);
    float4 b = *(const float4*)(in + i + 4);
    uint4 u;
    u.x = (unsigned)f32_to_bf16(a.x) | ((unsigned)f32_to_bf16(a.y) << 16);
    u.y = (unsigned)f32_to_bf16(a.z) | ((unsigned)f32_to_bf16(a.w) << 16);
    u.z = (unsigned)f32_to_bf16(b.x) | ((unsigned)f32_to_bf16(b.y) << 16);
    u.w = (unsigned)f32_to_bf16(b.z) | ((unsigned)f32_to_bf16(b.w) << 16);
    *(uint4*)(out + i) = u;
}

// -------- weight repack: (Cout,Cin,3,3) fp32 -> [tap][Cout][Cin] bf16 ------
__global__ __launch_bounds__(256)
void repack_w_bf16(const float* __restrict__ w, unsigned short* __restrict__ wT,
                   int Cout, int Cin)
{
    int idx = blockIdx.x * 256 + threadIdx.x;     // over 9*Cout*Cin
    if (idx >= 9 * Cout * Cin) return;
    int cin = idx % Cin;
    int t   = idx / Cin;
    int cout = t % Cout;
    int tap  = t / Cout;
    wT[idx] = f32_to_bf16(w[((size_t)cout * Cin + cin) * 9 + tap]);
}

#define MT 128          // Cout tile
#define NT 64           // pixel tile (one image row, W == 64)
#define KT 64           // K per stage: 2 WMMA k-chunks
#define LDA2 72         // ushorts per A row (64 + pad), 144B: 16B aligned
#define LDB2 72         // ushorts per B^T row

// Tap-major implicit GEMM conv3x3 (pad 1), bf16 in, bias+ReLU epilogue.
// in : (Nb, Cin, 64, 64) bf16;  wT: [tap][Cout][Cin] bf16
// out: (Nb, coutStride, 64, 64) at channel offset coutBase; bf16 or fp32.
// cshift: log2(Cin/64)
template <bool OUT_BF16>
__global__ __launch_bounds__(256, 2)
void conv3x3_taps_wmma(const unsigned short* __restrict__ in,
                       const unsigned short* __restrict__ wT,
                       const float* __restrict__ bias,
                       void* __restrict__ outP,
                       int Cin, int Cout, int coutBase, int coutStride, int cshift)
{
    __shared__ __align__(16) unsigned short As[2][MT * LDA2];  // A[m][k]
    __shared__ __align__(16) unsigned short Bs[2][NT * LDB2];  // B^T[n][k]

    const int tid  = threadIdx.x;
    const int wave = tid >> 5;
    const int lane = tid & 31;
    const int half = lane >> 4;      // K-half select per ISA fragment layout
    const int l16  = lane & 15;

    const int m0      = blockIdx.y * MT;        // Cout tile base
    const int pixBase = blockIdx.x * NT;        // one image row
    const int img     = pixBase >> 12;
    const int row     = (pixBase & 4095) >> 6;

    const int wm = (wave >> 1) * 32;            // wave M offset: 0/32/64/96
    const int wn = (wave & 1) * 32;             // wave N offset: 0/32

    const unsigned short* inImg = in + (size_t)img * Cin * 4096;
    const int CoutCin = Cout * Cin;

    // ---- per-thread staging maps (chunk-invariant) ----
    int aDst[4], aSrc[4];
    #pragma unroll
    for (int r = 0; r < 4; ++r) {
        int o  = tid + r * 256;                 // octet id, 1024 total
        int m  = o >> 3;                        // 8 octets per 64-wide row
        int ko = (o & 7) * 8;
        aDst[r] = m * LDA2 + ko;
        aSrc[r] = (m0 + m) * Cin + ko;
    }
    const int bN   = tid & 63;                  // B: one n column
    const int bKg  = (tid >> 6) * 16;           // 16 consecutive kk
    const int bDst = bN * LDB2 + bKg;

    // valid kh range for this row (block-uniform)
    const int khLo = (row == 0) ? 1 : 0;
    const int khHi = (row == 63) ? 1 : 2;
    const int nChunks = (khHi - khLo + 1) * 3 << cshift;
    const int cmask = (1 << cshift) - 1;

    v8f acc[2][2];
    #pragma unroll
    for (int a = 0; a < 2; ++a)
        #pragma unroll
        for (int b = 0; b < 2; ++b)
            acc[a][b] = (v8f){0.f,0.f,0.f,0.f,0.f,0.f,0.f,0.f};

    // ---- stage chunk ci into LDS buffer buf ----
    auto stage = [&](int buf, int ci) {
        int tIdx = ci >> cshift;                // uniform: which tap
        int cin0 = (ci & cmask) << 6;
        int dkh  = tIdx / 3;
        int kw   = tIdx - dkh * 3;
        int tap  = (khLo + dkh) * 3 + kw;
        int hy   = row + khLo + dkh - 1;        // guaranteed in [0,63]

        // A: 4 contiguous b128 loads -> b128 LDS stores
        const unsigned short* wTap = wT + (size_t)tap * CoutCin + cin0;
        unsigned short* Ab = &As[buf][0];
        #pragma unroll
        for (int r = 0; r < 4; ++r)
            *(uint4*)(Ab + aDst[r]) = *(const uint4*)(wTap + aSrc[r]);
        __builtin_prefetch(wTap + aSrc[0] + KT, 0, 3);  // speculative

        // B: 16 kk rows for one n, immediate-offset u16 loads, masked pack
        int wx = bN + kw - 1;
        bool ok = (unsigned)wx < 64u;
        unsigned msk = ok ? 0xFFFFFFFFu : 0u;
        const unsigned short* bp =
            inImg + ((size_t)bKg + cin0) * 4096 + hy * 64 + (ok ? wx : 0);
        unsigned t0  = bp[ 0 * 4096], t1  = bp[ 1 * 4096];
        unsigned t2  = bp[ 2 * 4096], t3  = bp[ 3 * 4096];
        unsigned t4  = bp[ 4 * 4096], t5  = bp[ 5 * 4096];
        unsigned t6  = bp[ 6 * 4096], t7  = bp[ 7 * 4096];
        unsigned t8  = bp[ 8 * 4096], t9  = bp[ 9 * 4096];
        unsigned t10 = bp[10 * 4096], t11 = bp[11 * 4096];
        unsigned t12 = bp[12 * 4096], t13 = bp[13 * 4096];
        unsigned t14 = bp[14 * 4096], t15 = bp[15 * 4096];
        uint4 p0, p1;
        p0.x = (t0  | (t1  << 16)) & msk;  p0.y = (t2  | (t3  << 16)) & msk;
        p0.z = (t4  | (t5  << 16)) & msk;  p0.w = (t6  | (t7  << 16)) & msk;
        p1.x = (t8  | (t9  << 16)) & msk;  p1.y = (t10 | (t11 << 16)) & msk;
        p1.z = (t12 | (t13 << 16)) & msk;  p1.w = (t14 | (t15 << 16)) & msk;
        unsigned short* Bb = &Bs[buf][bDst];
        *(uint4*)(Bb)     = p0;
        *(uint4*)(Bb + 8) = p1;
    };

    // ---- 8 WMMAs on LDS buffer buf ----
    auto compute = [&](int buf) {
        #pragma unroll
        for (int kc = 0; kc < 2; ++kc) {
            union Frag { v16bf v; uint4 q[2]; } af[2], bfm[2];
            #pragma unroll
            for (int a = 0; a < 2; ++a) {
                // A 16x32: lane<16 -> K {0..7,16..23}; lane>=16 -> {8..15,24..31}
                const unsigned short* p =
                    &As[buf][(wm + a * 16 + l16) * LDA2 + kc * 32 + half * 8];
                af[a].q[0] = *(const uint4*)(p);
                af[a].q[1] = *(const uint4*)(p + 16);
            }
            #pragma unroll
            for (int b = 0; b < 2; ++b) {
                // B 32x16: lane<16 -> K {0..7,16..23}; lane>=16 -> {8..15,24..31}
                const unsigned short* p =
                    &Bs[buf][(wn + b * 16 + l16) * LDB2 + kc * 32 + half * 8];
                bfm[b].q[0] = *(const uint4*)(p);
                bfm[b].q[1] = *(const uint4*)(p + 16);
            }
            #pragma unroll
            for (int a = 0; a < 2; ++a)
                #pragma unroll
                for (int b = 0; b < 2; ++b)
                    acc[a][b] = __builtin_amdgcn_wmma_f32_16x16x32_bf16(
                        false, af[a].v, false, bfm[b].v,
                        (short)0, acc[a][b], false, false);
        }
    };

    // ---- pipelined main loop: 1 barrier per stage, ping-pong LDS ----
    stage(0, 0);
    int buf = 0;
    #pragma unroll 1
    for (int ci = 0; ci < nChunks; ++ci) {
        __syncthreads();
        if (ci + 1 < nChunks) stage(buf ^ 1, ci + 1);
        compute(buf);
        buf ^= 1;
    }

    // ---- epilogue: bias + ReLU ----
    // C/D layout: VGPR r, lane<16 -> M=r, lane>=16 -> M=8+r; N = lane&15
    #pragma unroll
    for (int a = 0; a < 2; ++a) {
        #pragma unroll
        for (int b = 0; b < 2; ++b) {
            #pragma unroll
            for (int r = 0; r < 8; ++r) {
                int m    = wm + a * 16 + half * 8 + r;
                int n    = wn + b * 16 + l16;
                int cout = m0 + m;
                float v  = acc[a][b][r] + bias[cout];
                v = v > 0.f ? v : 0.f;
                size_t o = (((size_t)img * coutStride + coutBase + cout) << 12)
                           + row * 64 + n;
                if (OUT_BF16) ((unsigned short*)outP)[o] = f32_to_bf16(v);
                else          ((float*)outP)[o] = v;
            }
        }
    }
}

// 1x1 conv (256 -> 81) + bias + ReLU + channel softmax.
__global__ __launch_bounds__(256)
void conv1x1_softmax(const float* __restrict__ x,
                     const float* __restrict__ w3,
                     const float* __restrict__ b3,
                     float* __restrict__ kern)
{
    int p   = blockIdx.x * 256 + threadIdx.x;
    int img = p >> 12;
    int hw  = p & 4095;
    const float* xb = x + ((size_t)img * 256 << 12) + hw;

    float acc[81];
    #pragma unroll
    for (int o = 0; o < 81; ++o) acc[o] = b3[o];

    for (int c = 0; c < 256; ++c) {
        float xv = xb[(size_t)c << 12];        // coalesced across threads
        const float* wc = w3 + c;              // uniform -> scalar loads
        #pragma unroll
        for (int o = 0; o < 81; ++o)
            acc[o] = fmaf(xv, wc[o * 256], acc[o]);
    }

    float mx = 0.f;
    #pragma unroll
    for (int o = 0; o < 81; ++o) {
        acc[o] = acc[o] > 0.f ? acc[o] : 0.f;  // ReLU
        mx = acc[o] > mx ? acc[o] : mx;
    }
    float s = 0.f;
    #pragma unroll
    for (int o = 0; o < 81; ++o) { acc[o] = __expf(acc[o] - mx); s += acc[o]; }
    float inv = 1.f / s;

    float* kb = kern + ((size_t)img * 81 << 12) + hw;
    #pragma unroll
    for (int o = 0; o < 81; ++o) kb[(size_t)o << 12] = acc[o] * inv;
}

// Spatially-variant conv: out[b,c,h,w] = sum_{i,j} feats[b,c,h+i-4,w+j-4]*kern[b,i*9+j,h,w]
__global__ __launch_bounds__(256)
void svc_apply(const float* __restrict__ feats,
               const float* __restrict__ kern,
               float* __restrict__ out)
{
    __shared__ float ks[81 * 64];
    int bh  = blockIdx.x;          // b*64 + h
    int img = bh >> 6;
    int h   = bh & 63;
    int tid = threadIdx.x;

    const float* kb = kern + ((size_t)img * 81 << 12) + h * 64;
    for (int i = tid; i < 81 * 64; i += 256)
        ks[i] = kb[((size_t)(i >> 6) << 12) + (i & 63)];
    __syncthreads();

    int w     = tid & 63;
    int cLane = tid >> 6;
    for (int cg = 0; cg < 32; ++cg) {
        int c = blockIdx.y * 128 + cg * 4 + cLane;
        const float* fb = feats + (((size_t)img * 1024 + c) << 12);
        float acc = 0.f;
        #pragma unroll
        for (int i = 0; i < 9; ++i) {
            int hy = h + i - 4;
            bool rowOK = (unsigned)hy < 64u;
            const float* frow = fb + hy * 64;
            #pragma unroll
            for (int j = 0; j < 9; ++j) {
                int wx = w + j - 4;
                float fv = (rowOK && (unsigned)wx < 64u) ? frow[wx] : 0.f;
                acc = fmaf(fv, ks[(i * 9 + j) * 64 + w], acc);
            }
        }
        out[(((size_t)img * 1024 + c) << 12) + h * 64 + w] = acc;
    }
}

extern "C" void kernel_launch(void* const* d_in, const int* in_sizes, int n_in,
                              void* d_out, int out_size, void* d_ws, size_t ws_size,
                              hipStream_t stream) {
    (void)in_sizes; (void)n_in; (void)out_size; (void)ws_size;
    const float* cur = (const float*)d_in[0];   // (4,1024,64,64)
    const float* key = (const float*)d_in[1];   // (4,1024,64,64)
    const float* hi  = (const float*)d_in[2];   // (4,1024,64,64)
    const float* w1  = (const float*)d_in[3];   // (256,1024,3,3)
    const float* b1  = (const float*)d_in[4];
    const float* w2  = (const float*)d_in[5];   // (256,512,3,3)
    const float* b2  = (const float*)d_in[6];
    const float* w3  = (const float*)d_in[7];   // (81,256,1,1)
    const float* b3  = (const float*)d_in[8];
    float* out = (float*)d_out;                 // (4,1024,64,64)

    // workspace carve-out (~113 MB)
    char* ws = (char*)d_ws;
    const size_t nFrame = (size_t)4 * 1024 * 4096;    // 16.78M elems
    unsigned short* curb = (unsigned short*)ws;                 ws += nFrame * 2;
    unsigned short* keyb = (unsigned short*)ws;                 ws += nFrame * 2;
    unsigned short* x2b  = (unsigned short*)ws;                 ws += (size_t)4 * 512 * 4096 * 2;
    float*          x1   = (float*)ws;                          ws += (size_t)4 * 256 * 4096 * 4;
    float*          kn   = (float*)ws;                          ws += (size_t)4 * 81 * 4096 * 4;
    unsigned short* wT1  = (unsigned short*)ws;                 ws += (size_t)9 * 256 * 1024 * 2;
    unsigned short* wT2  = (unsigned short*)ws;

    dim3 blk(256);
    // activations -> bf16
    cvt_f32_bf16<<<dim3(nFrame / (256 * 8)), blk, 0, stream>>>(cur, curb);
    cvt_f32_bf16<<<dim3(nFrame / (256 * 8)), blk, 0, stream>>>(key, keyb);
    // weights -> [tap][Cout][Cin] bf16
    repack_w_bf16<<<dim3((9 * 256 * 1024 + 255) / 256), blk, 0, stream>>>(w1, wT1, 256, 1024);
    repack_w_bf16<<<dim3((9 * 256 * 512  + 255) / 256), blk, 0, stream>>>(w2, wT2, 256, 512);

    // conv_reduce on current/key frames -> channel-concat bf16 buffer
    conv3x3_taps_wmma<true ><<<dim3(4 * 64, 2), blk, 0, stream>>>(curb, wT1, b1, x2b, 1024, 256,   0, 512, 4);
    conv3x3_taps_wmma<true ><<<dim3(4 * 64, 2), blk, 0, stream>>>(keyb, wT1, b1, x2b, 1024, 256, 256, 512, 4);
    // conv2 on concat (512 -> 256), fp32 out
    conv3x3_taps_wmma<false><<<dim3(4 * 64, 2), blk, 0, stream>>>(x2b, wT2, b2, x1, 512, 256, 0, 256, 3);
    // 1x1 conv + ReLU + softmax
    conv1x1_softmax<<<dim3(64), blk, 0, stream>>>(x1, w3, b3, kn);
    // spatially-variant convolution
    svc_apply<<<dim3(256, 8), blk, 0, stream>>>(hi, kn, out);
}
